// EncoderBlock_8375186227595
// MI455X (gfx1250) — matrix operand order repaired and checked by reading, not compiled, not run on hardware
//
#include <hip/hip_runtime.h>
#include <hip/hip_bf16.h>

typedef __bf16 bf16_t;
typedef __attribute__((ext_vector_type(16))) __bf16 v16bf;
typedef __attribute__((ext_vector_type(8)))  __bf16 v8bf;
typedef __attribute__((ext_vector_type(8)))  float  v8f;

static constexpr int SEQ = 2048;
static constexpr int NB  = 2;
static constexpr int DM  = 1024;
static constexpr int NH  = 16;
static constexpr int DKH = 64;
static constexpr int DFF = 4096;
static constexpr int ROWS = NB * SEQ;   // 4096

// ---------------------------------------------------------------------------
// WMMA + fragment helpers
// ---------------------------------------------------------------------------
__device__ __forceinline__ v8f wmma_bf16(v16bf a, v16bf b, v8f c) {
  return __builtin_amdgcn_wmma_f32_16x16x32_bf16(false, a, false, b, (short)0, c,
                                                 false, false);
}

// A fragment (16x32): lane L -> row = r0+(L&15); 16 contiguous K values
// starting at k0 + (L>>4)*16 (matches B fragments). Works on global or LDS.
__device__ __forceinline__ v16bf load_a_frag(const bf16_t* __restrict__ base,
                                             int ld, int r0, int k0) {
  const int lane = threadIdx.x & 31;
  const bf16_t* p = base + (size_t)(r0 + (lane & 15)) * ld + (k0 + ((lane >> 4) << 4));
  v8bf lo = *(const v8bf*)p;
  v8bf hi = *(const v8bf*)(p + 8);
  v16bf r;
#pragma unroll
  for (int i = 0; i < 8; ++i) { r[i] = lo[i]; r[i + 8] = hi[i]; }
  return r;
}

// B fragment (32x16), B = W^T with W[N,K] row-major: lane holds col n0+(L&15),
// 16 contiguous K values starting at k0 + (L>>4)*16.
__device__ __forceinline__ v16bf load_b_frag(const bf16_t* __restrict__ base,
                                             int ld, int n0, int k0) {
  const int lane = threadIdx.x & 31;
  const bf16_t* p = base + (size_t)(n0 + (lane & 15)) * ld + (k0 + ((lane >> 4) << 4));
  v8bf lo = *(const v8bf*)p;
  v8bf hi = *(const v8bf*)(p + 8);
  v16bf r;
#pragma unroll
  for (int i = 0; i < 8; ++i) { r[i] = lo[i]; r[i + 8] = hi[i]; }
  return r;
}

// 16 contiguous bf16 at lane-specific pointer (LDS-resident B fragments).
__device__ __forceinline__ v16bf frag16(const bf16_t* p) {
  v8bf lo = *(const v8bf*)p;
  v8bf hi = *(const v8bf*)(p + 8);
  v16bf r;
#pragma unroll
  for (int i = 0; i < 8; ++i) { r[i] = lo[i]; r[i + 8] = hi[i]; }
  return r;
}

// ---------------------------------------------------------------------------
// CDNA5 async global->LDS copy (ASYNCcnt path), inline asm = toolchain-portable
// ---------------------------------------------------------------------------
__device__ __forceinline__ uint32_t lds_off(const void* p) {
  // generic pointer into LDS: low 32 bits are the workgroup-relative LDS offset
  return (uint32_t)(uintptr_t)p;
}
__device__ __forceinline__ void async_copy_b128(uint32_t dst_lds, const void* gsrc) {
  asm volatile("global_load_async_to_lds_b128 %0, %1, off"
               :: "v"(dst_lds), "v"(gsrc) : "memory");
}
__device__ __forceinline__ void async_wait0() {
  asm volatile("s_wait_asynccnt 0x0" ::: "memory");
}

// ---------------------------------------------------------------------------
// fp32 -> bf16 conversion
// ---------------------------------------------------------------------------
__global__ void f2bf_kernel(const float* __restrict__ in, bf16_t* __restrict__ out,
                            int n) {
  int i = blockIdx.x * blockDim.x + threadIdx.x;
  const int stride = gridDim.x * blockDim.x;
  for (; i < n; i += stride) out[i] = (bf16_t)in[i];
}

// ---------------------------------------------------------------------------
// LayerNorm (scalar alpha/bias, ddof=1, eps added to std), bf16 out
// ---------------------------------------------------------------------------
__global__ void ln_kernel(const float* __restrict__ x, const float* __restrict__ ap,
                          const float* __restrict__ bp, bf16_t* __restrict__ out) {
  const int row = blockIdx.x;
  const int t = threadIdx.x;  // 256
  const float* xr = x + (size_t)row * DM;
  float v[4];
  float s = 0.f;
#pragma unroll
  for (int i = 0; i < 4; ++i) { v[i] = xr[t + 256 * i]; s += v[i]; }
  __shared__ float red[8];
#pragma unroll
  for (int off = 16; off > 0; off >>= 1) s += __shfl_xor(s, off, 32);
  if ((t & 31) == 0) red[t >> 5] = s;
  __syncthreads();
  float mu = 0.f;
#pragma unroll
  for (int i = 0; i < 8; ++i) mu += red[i];
  mu *= (1.0f / DM);
  __syncthreads();
  float s2 = 0.f;
#pragma unroll
  for (int i = 0; i < 4; ++i) { float d = v[i] - mu; s2 += d * d; }
#pragma unroll
  for (int off = 16; off > 0; off >>= 1) s2 += __shfl_xor(s2, off, 32);
  if ((t & 31) == 0) red[t >> 5] = s2;
  __syncthreads();
  float var = 0.f;
#pragma unroll
  for (int i = 0; i < 8; ++i) var += red[i];
  var *= (1.0f / (DM - 1));
  const float stdv = sqrtf(var);
  const float a = ap[0], b = bp[0];
  const float inv = a / (stdv + 1e-6f);
  bf16_t* orow = out + (size_t)row * DM;
#pragma unroll
  for (int i = 0; i < 4; ++i) orow[t + 256 * i] = (bf16_t)((v[i] - mu) * inv + b);
}

// ---------------------------------------------------------------------------
// WMMA GEMM: C[M,N] = act(A @ W^T + bias) (+resid). Wave tile: 32x64.
// Block (8 waves) shares a 32-row A panel, double-buffered async-to-LDS so the
// DMA for panel i+1 overlaps the WMMAs of panel i.
// ---------------------------------------------------------------------------
template <int RELU, int RESID, int OUTBF>
__global__ void __launch_bounds__(256, 1)
gemm_kernel(const bf16_t* __restrict__ A, const bf16_t* __restrict__ W,
            const float* __restrict__ bias, const float* __restrict__ resid,
            void* __restrict__ outp, int M, int N, int K) {
  const int lane = threadIdx.x & 31;
  const int gw = blockIdx.x * 8 + (threadIdx.x >> 5);
  const int ntile = N >> 6;
  const int mt = gw / ntile;
  const int nt = gw - mt * ntile;
  const int r0 = mt << 5;
  const int c0 = nt << 6;

  __shared__ bf16_t At[2][32 * 64];       // double-buffered 4 KB panels
  const int t = threadIdx.x;
  const int srow = t >> 3;                // 0..31
  const int scol = (t & 7) << 3;          // 0..56
  const uint32_t at_dst0 = lds_off(&At[0][srow * 64 + scol]);
  const uint32_t at_dst1 = lds_off(&At[1][srow * 64 + scol]);
  const bf16_t* a_src = A + (size_t)(r0 + srow) * K + scol;

  v8f acc[2][4] = {};
  const int KT = K >> 6;

  async_copy_b128(at_dst0, a_src);
  async_wait0();
  __syncthreads();

  for (int it = 0; it < KT; ++it) {
    const int cur = it & 1;
    if (it + 1 < KT)
      async_copy_b128(cur ? at_dst0 : at_dst1, a_src + (size_t)(it + 1) * 64);
    const int k0 = it << 6;
    if (it + 2 <= KT)
      __builtin_prefetch(W + (size_t)(c0 + (lane & 15)) * K + k0 + 64, 0, 1);
#pragma unroll
    for (int ks = 0; ks < 64; ks += 32) {
      const v16bf a0 = load_a_frag(At[cur], 64, 0, ks);
      const v16bf a1 = load_a_frag(At[cur], 64, 16, ks);
#pragma unroll
      for (int j = 0; j < 4; ++j) {
        const v16bf b = load_b_frag(W, K, c0 + (j << 4), k0 + ks);
        acc[0][j] = wmma_bf16(a0, b, acc[0][j]);
        acc[1][j] = wmma_bf16(a1, b, acc[1][j]);
      }
    }
    if (it + 1 < KT) { async_wait0(); __syncthreads(); }
  }

  const int cl = lane & 15, g = lane >> 4;
#pragma unroll
  for (int rb = 0; rb < 2; ++rb)
#pragma unroll
    for (int j = 0; j < 4; ++j) {
      const int col = c0 + (j << 4) + cl;
      const float bb = bias[col];
#pragma unroll
      for (int v = 0; v < 8; ++v) {
        const int row = r0 + rb * 16 + (g << 3) + v;
        float val = acc[rb][j][v] + bb;
        if (RELU) val = fmaxf(val, 0.0f);
        if (RESID) val += resid[(size_t)row * N + col];
        if (OUTBF) ((bf16_t*)outp)[(size_t)row * N + col] = (bf16_t)val;
        else       ((float*)outp)[(size_t)row * N + col] = val;
      }
    }
}

// ---------------------------------------------------------------------------
// Flash attention. Block = 8 waves, one (b,h), 256 query rows (32 per wave).
// Double-buffered K (async-to-LDS) and transposed-V panels. K B-fragments are
// loaded once per tile and shared by both query row-blocks (8 up-front WMMAs).
// ---------------------------------------------------------------------------
__global__ void __launch_bounds__(256, 1)
attn_kernel(const bf16_t* __restrict__ Q, const bf16_t* __restrict__ K,
            const bf16_t* __restrict__ V, const int* __restrict__ mask,
            bf16_t* __restrict__ O) {
  const int b = blockIdx.z, h = blockIdx.y;
  const int wave = threadIdx.x >> 5;
  const int lane = threadIdx.x & 31;
  const int cl = lane & 15, g = lane >> 4;
  const int q0 = blockIdx.x * 256 + wave * 32;

  const bf16_t* Qb = Q + (size_t)b * SEQ * DM + h * DKH;
  const bf16_t* Kb = K + (size_t)b * SEQ * DM + h * DKH;
  const bf16_t* Vb = V + (size_t)b * SEQ * DM + h * DKH;
  const int* mb = mask + b * SEQ;

  __shared__ bf16_t Kt[2][32 * 64];       // key panels, row-major      2x4 KB
  __shared__ bf16_t VtT[2][64 * 32];      // transposed V: [d][key]     2x4 KB
  __shared__ bf16_t Pt[8][16 * 32];       // per-wave P transpose        8 KB

  const int t = threadIdx.x;
  const int srow = t >> 3;                // staged key row 0..31
  const int scol = (t & 7) << 3;          // staged dk col 0..56
  const uint32_t kt_dst0 = lds_off(&Kt[0][srow * 64 + scol]);
  const uint32_t kt_dst1 = lds_off(&Kt[1][srow * 64 + scol]);

  auto stage = [&](int buf, int kt) {
    async_copy_b128(buf ? kt_dst1 : kt_dst0,
                    Kb + (size_t)(kt + srow) * DM + scol);
    v8bf vv = *(const v8bf*)(Vb + (size_t)(kt + srow) * DM + scol);
#pragma unroll
    for (int i = 0; i < 8; ++i) VtT[buf][(scol + i) * 32 + srow] = vv[i];
  };

  stage(0, 0);

  v16bf aq[2][2];
#pragma unroll
  for (int rb = 0; rb < 2; ++rb) {
    aq[rb][0] = load_a_frag(Qb, DM, q0 + rb * 16, 0);
    aq[rb][1] = load_a_frag(Qb, DM, q0 + rb * 16, 32);
  }

  v8f acc[2][4] = {};
  float m[2][8], l[2][8];
#pragma unroll
  for (int rb = 0; rb < 2; ++rb)
#pragma unroll
    for (int v = 0; v < 8; ++v) { m[rb][v] = -1e30f; l[rb][v] = 0.f; }

  async_wait0();
  __syncthreads();

  const int NT = SEQ / 32;                // 64 key tiles
  for (int it = 0; it < NT; ++it) {
    const int cur = it & 1;
    if (it + 1 < NT) stage(cur ^ 1, (it + 1) * 32);
    const int kt = it * 32;
    const bf16_t* Kc = Kt[cur];
    const bf16_t* Vc = VtT[cur];

    const bool keep0 = (mb[kt + cl] != 0);
    const bool keep1 = (mb[kt + 16 + cl] != 0);

    // ---- shared K B-fragments (identical for both row-blocks) ----
    const v16bf kf0a = frag16(&Kc[cl * 64 + (g << 4)]);
    const v16bf kf0b = frag16(&Kc[cl * 64 + 32 + (g << 4)]);
    const v16bf kf1a = frag16(&Kc[(16 + cl) * 64 + (g << 4)]);
    const v16bf kf1b = frag16(&Kc[(16 + cl) * 64 + 32 + (g << 4)]);

    // ---- scores for both row-blocks: 8 independent WMMAs up front ----
    v8f s[2][2];
#pragma unroll
    for (int rb = 0; rb < 2; ++rb) {
      v8f z0 = {};
      z0 = wmma_bf16(aq[rb][0], kf0a, z0);
      z0 = wmma_bf16(aq[rb][1], kf0b, z0);
      v8f z1 = {};
      z1 = wmma_bf16(aq[rb][0], kf1a, z1);
      z1 = wmma_bf16(aq[rb][1], kf1b, z1);
#pragma unroll
      for (int v = 0; v < 8; ++v) {
        const float sc0 = z0[v] * 0.125f;   // 1/sqrt(64)
        const float sc1 = z1[v] * 0.125f;
        z0[v] = keep0 ? sc0 : -1e9f;
        z1[v] = keep1 ? sc1 : -1e9f;
      }
      s[rb][0] = z0;
      s[rb][1] = z1;
    }

#pragma unroll
    for (int rb = 0; rb < 2; ++rb) {
      // ---- online softmax (rows striped across 16-lane halves) ----
      float rmax[8], rsum[8], alf[8];
#pragma unroll
      for (int v = 0; v < 8; ++v) rmax[v] = fmaxf(s[rb][0][v], s[rb][1][v]);
#pragma unroll
      for (int off = 1; off < 16; off <<= 1)
#pragma unroll
        for (int v = 0; v < 8; ++v)
          rmax[v] = fmaxf(rmax[v], __shfl_xor(rmax[v], off, 32));
#pragma unroll
      for (int v = 0; v < 8; ++v) {
        const float mn = fmaxf(m[rb][v], rmax[v]);
        alf[v] = __expf(m[rb][v] - mn);
        m[rb][v] = mn;
        const float p0 = __expf(s[rb][0][v] - mn);
        const float p1 = __expf(s[rb][1][v] - mn);
        s[rb][0][v] = p0; s[rb][1][v] = p1;
        rsum[v] = p0 + p1;
      }
#pragma unroll
      for (int off = 1; off < 16; off <<= 1)
#pragma unroll
        for (int v = 0; v < 8; ++v) rsum[v] += __shfl_xor(rsum[v], off, 32);
#pragma unroll
      for (int v = 0; v < 8; ++v) l[rb][v] = l[rb][v] * alf[v] + rsum[v];
#pragma unroll
      for (int tt = 0; tt < 4; ++tt)
#pragma unroll
        for (int v = 0; v < 8; ++v) acc[rb][tt][v] *= alf[v];

      // ---- P: C layout -> A layout through per-wave LDS region ----
#pragma unroll
      for (int j = 0; j < 2; ++j)
#pragma unroll
        for (int v = 0; v < 8; ++v)
          Pt[wave][(v + 8 * g) * 32 + j * 16 + cl] = (bf16_t)s[rb][j][v];
      v16bf pa;
      {
        const bf16_t* pp = &Pt[wave][cl * 32 + (g << 4)];
#pragma unroll
        for (int i = 0; i < 8; ++i) { pa[i] = pp[i]; pa[i + 8] = pp[i + 8]; }
      }
      // ---- P @ V : B frags contiguous from transposed V panel ----
#pragma unroll
      for (int tt = 0; tt < 4; ++tt) {
        const v16bf bv = frag16(&Vc[(tt * 16 + cl) * 32 + (g << 4)]);
        acc[rb][tt] = wmma_bf16(pa, bv, acc[rb][tt]);
      }
    }
    if (it + 1 < NT) { async_wait0(); __syncthreads(); }
  }

  bf16_t* Ob = O + (size_t)b * SEQ * DM + h * DKH;
#pragma unroll
  for (int rb = 0; rb < 2; ++rb)
#pragma unroll
    for (int tt = 0; tt < 4; ++tt)
#pragma unroll
      for (int v = 0; v < 8; ++v) {
        const int row = q0 + rb * 16 + (g << 3) + v;
        Ob[(size_t)row * DM + tt * 16 + cl] = (bf16_t)(acc[rb][tt][v] / l[rb][v]);
      }
}

// ---------------------------------------------------------------------------
// Launch
// ---------------------------------------------------------------------------
extern "C" void kernel_launch(void* const* d_in, const int* in_sizes, int n_in,
                              void* d_out, int out_size, void* d_ws, size_t ws_size,
                              hipStream_t stream) {
  const float* x    = (const float*)d_in[0];
  const int*   mask = (const int*)d_in[1];
  const float* wq = (const float*)d_in[2];  const float* bq = (const float*)d_in[3];
  const float* wk = (const float*)d_in[4];  const float* bk = (const float*)d_in[5];
  const float* wv = (const float*)d_in[6];  const float* bv = (const float*)d_in[7];
  const float* wo = (const float*)d_in[8];  const float* bo = (const float*)d_in[9];
  const float* w1 = (const float*)d_in[10]; const float* b1 = (const float*)d_in[11];
  const float* w2 = (const float*)d_in[12]; const float* b2 = (const float*)d_in[13];
  const float* ln1a = (const float*)d_in[14]; const float* ln1b = (const float*)d_in[15];
  const float* ln2a = (const float*)d_in[16]; const float* ln2b = (const float*)d_in[17];

  char* ws = (char*)d_ws;
  size_t off = 0;
  auto take = [&](size_t elems, size_t esz) -> void* {
    void* p = ws + off;
    off = (off + elems * esz + 255) & ~(size_t)255;
    return p;
  };
  bf16_t* wq_b = (bf16_t*)take((size_t)DM * DM, 2);
  bf16_t* wk_b = (bf16_t*)take((size_t)DM * DM, 2);
  bf16_t* wv_b = (bf16_t*)take((size_t)DM * DM, 2);
  bf16_t* wo_b = (bf16_t*)take((size_t)DM * DM, 2);
  bf16_t* w1_b = (bf16_t*)take((size_t)DFF * DM, 2);
  bf16_t* w2_b = (bf16_t*)take((size_t)DM * DFF, 2);
  bf16_t* xn   = (bf16_t*)take((size_t)ROWS * DM, 2);
  bf16_t* qb   = (bf16_t*)take((size_t)ROWS * DM, 2);
  bf16_t* kb   = (bf16_t*)take((size_t)ROWS * DM, 2);
  bf16_t* vb   = (bf16_t*)take((size_t)ROWS * DM, 2);
  bf16_t* attn = (bf16_t*)take((size_t)ROWS * DM, 2);
  float*  x1   = (float*) take((size_t)ROWS * DM, 4);
  bf16_t* hbuf = (bf16_t*)take((size_t)ROWS * DFF, 2);

  // 1) weights -> bf16
  f2bf_kernel<<<2048, 256, 0, stream>>>(wq, wq_b, DM * DM);
  f2bf_kernel<<<2048, 256, 0, stream>>>(wk, wk_b, DM * DM);
  f2bf_kernel<<<2048, 256, 0, stream>>>(wv, wv_b, DM * DM);
  f2bf_kernel<<<2048, 256, 0, stream>>>(wo, wo_b, DM * DM);
  f2bf_kernel<<<2048, 256, 0, stream>>>(w1, w1_b, DFF * DM);
  f2bf_kernel<<<2048, 256, 0, stream>>>(w2, w2_b, DM * DFF);

  // 2) LN1
  ln_kernel<<<ROWS, 256, 0, stream>>>(x, ln1a, ln1b, xn);

  // 3) Q,K,V projections: waves = (M/32)*(N/64), 8 waves/block
  const int blkProj = (ROWS / 32) * (DM / 64) / 8;   // 256
  gemm_kernel<0, 0, 1><<<blkProj, 256, 0, stream>>>(xn, wq_b, bq, nullptr, qb, ROWS, DM, DM);
  gemm_kernel<0, 0, 1><<<blkProj, 256, 0, stream>>>(xn, wk_b, bk, nullptr, kb, ROWS, DM, DM);
  gemm_kernel<0, 0, 1><<<blkProj, 256, 0, stream>>>(xn, wv_b, bv, nullptr, vb, ROWS, DM, DM);

  // 4) attention: 8 blocks of 256 query rows per (b,h)
  attn_kernel<<<dim3(SEQ / 256, NH, NB), 256, 0, stream>>>(qb, kb, vb, mask, attn);

  // 5) output projection + residual -> x1 (f32)
  gemm_kernel<0, 1, 0><<<blkProj, 256, 0, stream>>>(attn, wo_b, bo, x, x1, ROWS, DM, DM);

  // 6) LN2
  ln_kernel<<<ROWS, 256, 0, stream>>>(x1, ln2a, ln2b, xn);

  // 7) FFN1 (ReLU) -> h (bf16)
  const int blkFF1 = (ROWS / 32) * (DFF / 64) / 8;   // 1024
  gemm_kernel<1, 0, 1><<<blkFF1, 256, 0, stream>>>(xn, w1_b, b1, nullptr, hbuf, ROWS, DFF, DM);

  // 8) FFN2 + residual -> d_out (f32)
  gemm_kernel<0, 1, 0><<<blkProj, 256, 0, stream>>>(hbuf, w2_b, b2, x1, d_out, ROWS, DM, DFF);
}
